// RPEMultiHeadAttention_27066883899425
// MI455X (gfx1250) — matrix operand order repaired and checked by reading, not compiled, not run on hardware
//
#include <hip/hip_runtime.h>
#include <cmath>

// Problem constants (match reference)
constexpr int B_ = 4;
constexpr int N_ = 1024;
constexpr int D_ = 512;
constexpr int P_ = 10;
constexpr int H_ = 8;
constexpr int C_ = 64;   // D/H

typedef __attribute__((ext_vector_type(16))) _Float16 v16h;
typedef __attribute__((ext_vector_type(8)))  _Float16 v8h;
typedef __attribute__((ext_vector_type(8)))  float    v8f;

union Frag16 { v16h v; v8h h[2]; };

// A-matrix fragment (16xK tile, this lane = row `mrow`, K-window of 32):
// lane<16 holds K = {0..7, 16..23}, lane>=16 holds K = {8..15, 24..31}
// -> two contiguous 16B chunks at +8*half and +16+8*half.
__device__ __forceinline__ v16h load_afrag(const _Float16* rowbase, int half) {
  Frag16 f;
  f.h[0] = *(const v8h*)(rowbase + 8 * half);
  f.h[1] = *(const v8h*)(rowbase + 16 + 8 * half);
  return f.v;
}

// B-matrix fragment (Kx16 tile, this lane = col `ncol`, K-window of 32):
// lane<16 holds K = {0..7, 16..23}, lane>=16 holds K = {8..15, 24..31}
// -> two contiguous 16B chunks at +8*half and +16+8*half.
__device__ __forceinline__ v16h load_bfrag(const _Float16* colbase, int half) {
  Frag16 f;
  f.h[0] = *(const v8h*)(colbase + 8 * half);
  f.h[1] = *(const v8h*)(colbase + 16 + 8 * half);
  return f.v;
}

__device__ __forceinline__ v8f wmma_f16(v16h a, v16h b, v8f c) {
  return __builtin_amdgcn_wmma_f32_16x16x32_f16(
      false, a, false, b, (short)0, c, false, false);
}

// ---------------------------------------------------------------------------
// Kernel 1: projection  out = heads(X(b,n,K) @ W(K,D) + bias), f16 output.
// TRANSPOSED=false -> (B,H,N,C) layout (q/k/p: C is the GEMM K dim later)
// TRANSPOSED=true  -> (B,H,C,N) layout (v/vp: N is the GEMM K dim later)
// block = 256 (8 waves): 16 rows x 128 cols; A and W^T tiles staged in LDS.
// ---------------------------------------------------------------------------
template <bool TRANSPOSED>
__global__ __launch_bounds__(256) void proj_kernel(
    const float* __restrict__ X, const float* __restrict__ W,
    const float* __restrict__ bias, _Float16* __restrict__ out, int K)
{
  __shared__ _Float16 As[16][40];    // 16 rows x 32 K   (80B rows, 16B aligned)
  __shared__ _Float16 Ws[128][40];   // 128 cols x 32 K  (K-contiguous per col)

  const int b    = blockIdx.z;
  const int n0   = blockIdx.x * 16;
  const int w    = threadIdx.x >> 5;
  const int lane = threadIdx.x & 31;
  const int half = lane >> 4;
  const int mrow = lane & 15;
  const int ncol = lane & 15;
  const int cbase = blockIdx.y * 128;

  const int ar = threadIdx.x >> 4;            // A staging: row 0..15
  const int ak = (threadIdx.x & 15) * 2;      //            k pair 0..30
  const size_t xrow = ((size_t)b * N_ + n0 + ar) * K;

  v8f acc = {};
  const int Kr = (K + 31) & ~31;
  for (int k0 = 0; k0 < Kr; k0 += 32) {
    // stage A tile (16x32), f32->f16, coalesced float2 per thread
    {
      int kg = k0 + ak;
      float2 t2 = make_float2(0.f, 0.f);
      if (kg + 1 < K)      t2 = *(const float2*)&X[xrow + kg];
      else if (kg < K)     t2.x = X[xrow + kg];
      As[ar][ak]     = (_Float16)t2.x;
      As[ar][ak + 1] = (_Float16)t2.y;
    }
    // stage W tile transposed (32 K x 128 cols -> Ws[col][k])
#pragma unroll
    for (int i = 0; i < 16; ++i) {
      int e = threadIdx.x + i * 256;
      int kk = e >> 7, col = e & 127;
      int kg = k0 + kk;
      float v = (kg < K) ? W[(size_t)kg * D_ + cbase + col] : 0.0f;
      Ws[col][kk] = (_Float16)v;
    }
    __syncthreads();

    v16h a = load_afrag(&As[mrow][0], half);
    v16h bf = load_bfrag(&Ws[w * 16 + ncol][0], half);
    acc = wmma_f16(a, bf, acc);
    __syncthreads();
  }

  const size_t bhbase = (size_t)b * H_;  // head index derived per column
#pragma unroll
  for (int i = 0; i < 8; ++i) {
    int row = i + 8 * half;              // n within tile
    int col = cbase + w * 16 + ncol;     // d within D
    float v = acc[i] + bias[col];
    int h = col >> 6, c = col & 63;      // C_ = 64
    size_t bh = bhbase + h;
    if (TRANSPOSED)
      out[(bh * C_ + c) * N_ + (n0 + row)] = (_Float16)v;
    else
      out[(bh * N_ + (n0 + row)) * C_ + c] = (_Float16)v;
  }
}

// ---------------------------------------------------------------------------
// Kernel 2: fused scores + softmax.
// block = 256 (8 waves) owns 16 query rows of one (b,h). Wave w computes the
// 8 m-tiles {w*16 + it*128}; the full 16x1024 score strip lives in registers
// (8 x v8f per lane). Softmax reduces via LDS, attn written exactly once.
// ---------------------------------------------------------------------------
__global__ __launch_bounds__(256) void score_softmax_kernel(
    const _Float16* __restrict__ qh, const _Float16* __restrict__ kh,
    const _Float16* __restrict__ ph, const float* __restrict__ af,
    const float* __restrict__ kw, const int* __restrict__ mask,
    float* __restrict__ attn)
{
  __shared__ float part[16][132];   // per-row partials from 128 lane-slots
  __shared__ float red[16][17];
  __shared__ float rowmax[16];
  __shared__ float rowsum[16];

  const int b = blockIdx.z, h = blockIdx.y;
  const int n0   = blockIdx.x * 16;
  const int w    = threadIdx.x >> 5;
  const int lane = threadIdx.x & 31;
  const int half = lane >> 4, mrow = lane & 15, ncol = lane & 15;
  const size_t bh = (size_t)b * H_ + h;
  const int team = threadIdx.x >> 4;   // reduction: row 0..15
  const int lt   = threadIdx.x & 15;

  // A fragments (q rows) are invariant over m: hoist both K-windows.
  const _Float16* qrow = qh + (bh * N_ + n0 + mrow) * C_;
  const v16h a0 = load_afrag(qrow, half);
  const v16h a1 = load_afrag(qrow + 32, half);

  v8f acc[8];
#pragma unroll
  for (int it = 0; it < 8; ++it) {
    const int m0 = w * 16 + it * 128;
    const _Float16* kcol = kh + (bh * N_ + m0 + ncol) * C_;
    const _Float16* pcol = ph + (bh * N_ + m0 + ncol) * C_;
    if (it < 7) {
      __builtin_prefetch(kcol + 128 * C_, 0, 1);
      __builtin_prefetch(pcol + 128 * C_, 0, 1);
    }
    v8f t = {};
    t = wmma_f16(a0, load_bfrag(kcol, half), t);
    t = wmma_f16(a1, load_bfrag(kcol + 32, half), t);
    t = wmma_f16(a0, load_bfrag(pcol, half), t);
    t = wmma_f16(a1, load_bfrag(pcol + 32, half), t);

    const int gm = m0 + ncol;
    const float kwv = kw[(size_t)b * N_ + gm] * 0.125f;  // fold 1/sqrt(64)
    const bool msk = mask[(size_t)b * N_ + gm] != 0;
#pragma unroll
    for (int i = 0; i < 8; ++i) {
      int gn = n0 + i + 8 * half;
      float v = t[i] * kwv * af[((size_t)b * N_ + gn) * N_ + gm];
      t[i] = msk ? -INFINITY : v;
    }
    acc[it] = t;
  }

  // ---- row max ----
  float pm[8];
#pragma unroll
  for (int i = 0; i < 8; ++i) {
    float m = acc[0][i];
#pragma unroll
    for (int it = 1; it < 8; ++it) m = fmaxf(m, acc[it][i]);
    pm[i] = m;
  }
#pragma unroll
  for (int i = 0; i < 8; ++i) part[i + 8 * half][w * 16 + ncol] = pm[i];
  __syncthreads();
  {
    float tm = part[team][lt];
#pragma unroll
    for (int j = 1; j < 8; ++j) tm = fmaxf(tm, part[team][lt + 16 * j]);
    red[team][lt] = tm;
  }
  __syncthreads();
  if (lt == 0) {
    float mm = red[team][0];
#pragma unroll
    for (int j = 1; j < 16; ++j) mm = fmaxf(mm, red[team][j]);
    rowmax[team] = mm;
  }
  __syncthreads();

  // ---- exp + row sum ----
  float rmax[8], ps[8];
#pragma unroll
  for (int i = 0; i < 8; ++i) { rmax[i] = rowmax[i + 8 * half]; ps[i] = 0.f; }
#pragma unroll
  for (int it = 0; it < 8; ++it) {
#pragma unroll
    for (int i = 0; i < 8; ++i) {
      float e = __expf(acc[it][i] - rmax[i]);
      acc[it][i] = e;
      ps[i] += e;
    }
  }
#pragma unroll
  for (int i = 0; i < 8; ++i) part[i + 8 * half][w * 16 + ncol] = ps[i];
  __syncthreads();
  {
    float ts = part[team][lt];
#pragma unroll
    for (int j = 1; j < 8; ++j) ts += part[team][lt + 16 * j];
    red[team][lt] = ts;
  }
  __syncthreads();
  if (lt == 0) {
    float s = red[team][0];
#pragma unroll
    for (int j = 1; j < 16; ++j) s += red[team][j];
    rowsum[team] = s;
  }
  __syncthreads();

  // ---- normalize + single write of attn ----
  float rinv[8];
#pragma unroll
  for (int i = 0; i < 8; ++i) rinv[i] = 1.0f / rowsum[i + 8 * half];
#pragma unroll
  for (int it = 0; it < 8; ++it) {
    const int gm = w * 16 + it * 128 + ncol;
#pragma unroll
    for (int i = 0; i < 8; ++i) {
      int gn = n0 + i + 8 * half;
      attn[(bh * N_ + gn) * N_ + gm] = acc[it][i] * rinv[i];
    }
  }
}

// ---------------------------------------------------------------------------
// Kernel 3: hidden = attn @ v, pos = attn @ vp  (v, vp stored transposed).
// block = 256 (8 waves): waves 0-3 -> hidden c-tiles 0..3, 4-7 -> pos.
// attn 16x32 f32 tile staged to LDS as f16, shared 8 ways; B fragments are
// contiguous b128 loads from the (B,H,C,N) transposed layout.
// ---------------------------------------------------------------------------
__global__ __launch_bounds__(256) void av_kernel(
    const float* __restrict__ attn, const _Float16* __restrict__ vT,
    const _Float16* __restrict__ vpT, float* __restrict__ hidden,
    float* __restrict__ pos)
{
  __shared__ _Float16 Ah[16][40];

  const int bh = blockIdx.z;
  const int b = bh >> 3, h = bh & 7;   // H_ = 8
  const int n0   = blockIdx.x * 16;
  const int w    = threadIdx.x >> 5;
  const int lane = threadIdx.x & 31;
  const int half = lane >> 4, mrow = lane & 15, ncol = lane & 15;
  const int c0 = (w & 3) * 16;
  const bool doPos = (w >= 4);
  const _Float16* src = doPos ? vpT : vT;

  const float* arow = attn + ((size_t)bh * N_ + n0) * N_;
  const _Float16* bbase = src + ((size_t)bh * C_ + c0 + ncol) * N_;

  const int ar = threadIdx.x >> 4;         // staging row 0..15
  const int ak = (threadIdx.x & 15) * 2;   // staging k pair

  v8f acc = {};
  for (int m0 = 0; m0 < N_; m0 += 32) {
    {
      const float2 t2 = *(const float2*)&arow[(size_t)ar * N_ + m0 + ak];
      Ah[ar][ak]     = (_Float16)t2.x;
      Ah[ar][ak + 1] = (_Float16)t2.y;
    }
    if (m0 + 32 < N_) __builtin_prefetch(&arow[(size_t)ar * N_ + m0 + 32 + ak], 0, 1);
    __syncthreads();

    v16h a  = load_afrag(&Ah[mrow][0], half);
    v16h bf = load_bfrag(bbase + m0, half);
    acc = wmma_f16(a, bf, acc);
    __syncthreads();
  }

  float* dst = doPos ? pos : hidden;
#pragma unroll
  for (int i = 0; i < 8; ++i) {
    int gn = n0 + i + 8 * half;
    dst[((size_t)b * N_ + gn) * D_ + h * C_ + c0 + ncol] = acc[i];
  }
}

// ---------------------------------------------------------------------------
extern "C" void kernel_launch(void* const* d_in, const int* in_sizes, int n_in,
                              void* d_out, int out_size, void* d_ws, size_t ws_size,
                              hipStream_t stream) {
  const float* input_q = (const float*)d_in[0];
  const float* input_k = (const float*)d_in[1];
  const float* input_v = (const float*)d_in[2];
  const float* embed   = (const float*)d_in[3];
  const float* kweight = (const float*)d_in[4];
  const int*   kmask   = (const int*)d_in[5];
  const float* afac    = (const float*)d_in[6];
  const float* Wq = (const float*)d_in[7];   const float* bq  = (const float*)d_in[8];
  const float* Wk = (const float*)d_in[9];   const float* bk  = (const float*)d_in[10];
  const float* Wv = (const float*)d_in[11];  const float* bv  = (const float*)d_in[12];
  const float* Wp = (const float*)d_in[13];  const float* bp  = (const float*)d_in[14];
  const float* Wvp = (const float*)d_in[15]; const float* bvp = (const float*)d_in[16];

  // workspace: 5 f16 tensors of B*H*N*C (4 MB each, 20 MB total)
  const size_t seg = (size_t)B_ * H_ * N_ * C_;
  _Float16* qh  = (_Float16*)d_ws;
  _Float16* kh  = qh + seg;
  _Float16* ph  = kh + seg;
  _Float16* vT  = ph + seg;   // (B,H,C,N)
  _Float16* vpT = vT + seg;   // (B,H,C,N)

  // outputs, flat: hidden (B,N,D) | attn (B,H,N,N) | pos (B,N,D)
  float* hidden = (float*)d_out;
  float* attn   = hidden + (size_t)B_ * N_ * D_;
  float* pos    = attn + (size_t)B_ * H_ * N_ * N_;

  dim3 blk(256);
  dim3 gproj(N_ / 16, D_ / 128, B_);
  proj_kernel<false><<<gproj, blk, 0, stream>>>(input_q, Wq, bq, qh, D_);
  proj_kernel<false><<<gproj, blk, 0, stream>>>(input_k, Wk, bk, kh, D_);
  proj_kernel<false><<<gproj, blk, 0, stream>>>(embed, Wp, bp, ph, P_);
  proj_kernel<true><<<gproj, blk, 0, stream>>>(input_v, Wv, bv, vT, D_);
  proj_kernel<true><<<gproj, blk, 0, stream>>>(embed, Wvp, bvp, vpT, P_);

  score_softmax_kernel<<<dim3(N_ / 16, H_, B_), blk, 0, stream>>>(
      qh, kh, ph, afac, kweight, kmask, attn);

  av_kernel<<<dim3(N_ / 16, 1, B_ * H_), blk, 0, stream>>>(
      attn, vT, vpT, hidden, pos);
}
